// ExpertsFeedForwardNetwork_87806311399467
// MI455X (gfx1250) — matrix operand order repaired and checked by reading, not compiled, not run on hardware
//
#include <hip/hip_runtime.h>
#include <hip/hip_bf16.h>

// ---------------- problem constants ----------------
#define BB 2
#define SS 2048
#define HH 1024          // hidden
#define FF 4096          // ffn
#define EE 8             // experts
#define KK 2             // top-k
#define TT (BB*SS)       // 4096 tokens
#define MAXT (TT*KK)     // worst-case slots per expert = 8192
#define TM 32            // token tile per block
#define FC 1024          // F columns per block (z dim)
#define NZ (FF/FC)       // 4

// ---------------- LDS layout (bytes) ----------------
#define XP   1040        // Xs pitch (bf16 elems): 1024 + pad
#define W1P  1034        // W1 panel pitch [f][k]: 1024 + pad (even)
#define W2P  72          // W2 panel pitch [h][k]: 64 + pad (even)
#define HP   72          // h  panel pitch [m][f]: 64 + pad (even)
#define XS_OFF   0
#define WS_OFF   66560                    // 32*1040*2
#define HS_OFF   214016                   // WS_OFF + 1024*72*2 (>= 64*1034*2)
#define WROW_OFF 218624                   // HS_OFF + 32*72*2
#define TOK_OFF  218752
#define SMEM_BYTES 218880

typedef __attribute__((ext_vector_type(16))) __bf16 v16bf;
typedef __attribute__((ext_vector_type(8)))  float  v8f;

union ABFrag { unsigned int u[8]; v16bf v; };

__device__ __forceinline__ unsigned short f2bf(float f) {
  unsigned int u = __float_as_uint(f);
  u += 0x7FFFu + ((u >> 16) & 1u);      // round-to-nearest-even
  return (unsigned short)(u >> 16);
}

__device__ __forceinline__ float gelu_tanh(float v) {
  float c = 0.7978845608028654f * (v + 0.044715f * v * v * v);
  return 0.5f * v * (1.0f + tanhf(c));
}

// A fragment (16x32 bf16): lane = row (lrow), half selects K group.
// rowPtr already includes m*pitch + k0.
__device__ __forceinline__ v16bf loadA(const unsigned short* rowPtr, int half) {
  ABFrag a;
  const unsigned short* p0 = rowPtr + half * 8;
#pragma unroll
  for (int r = 0; r < 4; ++r) a.u[r]     = *(const unsigned int*)(p0 + 2 * r);
#pragma unroll
  for (int r = 0; r < 4; ++r) a.u[4 + r] = *(const unsigned int*)(p0 + 16 + 2 * r);
  return a.v;
}

// B fragment (32x16 bf16) from a [n][k]-transposed panel: lane = col, half = K half.
// colPtr already includes n*pitch + k0.
__device__ __forceinline__ v16bf loadB(const unsigned short* colPtr, int half) {
  ABFrag b;
  const unsigned short* p0 = colPtr + half * 8;
#pragma unroll
  for (int r = 0; r < 4; ++r) b.u[r]     = *(const unsigned int*)(p0 + 2 * r);
#pragma unroll
  for (int r = 0; r < 4; ++r) b.u[4 + r] = *(const unsigned int*)(p0 + 16 + 2 * r);
  return b.v;
}

// ---------------- routing: build per-expert token lists ----------------
__global__ void moe_route(const float* __restrict__ probs,
                          const int*   __restrict__ experts,
                          int* __restrict__ cnt,
                          int* __restrict__ tokid,
                          float* __restrict__ tokw) {
  int i = blockIdx.x * blockDim.x + threadIdx.x;     // over K*T = 8192 slots
  if (i >= KK * TT) return;
  int e = experts[i];
  float p = probs[i];
  int pos = atomicAdd(&cnt[e], 1);
  tokid[e * MAXT + pos] = i & (TT - 1);              // token index
  tokw [e * MAXT + pos] = p;
}

// ---------------- fused gathered FFN ----------------
__global__ __launch_bounds__(256)
void moe_ffn(const float* __restrict__ x,  const float* __restrict__ w1,
             const float* __restrict__ b1, const float* __restrict__ w2,
             const float* __restrict__ b2, const int*   __restrict__ cnt,
             const int*   __restrict__ tokid, const float* __restrict__ tokw,
             float* __restrict__ out) {
  extern __shared__ char smem[];
  unsigned short* Xs  = (unsigned short*)(smem + XS_OFF);   // [32][XP] bf16
  unsigned short* Ws  = (unsigned short*)(smem + WS_OFF);   // W1 [64][W1P] | W2 [1024][W2P]
  unsigned short* Hs  = (unsigned short*)(smem + HS_OFF);   // [32][HP] bf16
  float*          wrS = (float*)(smem + WROW_OFF);          // [32]
  int*            tkS = (int*)  (smem + TOK_OFF);           // [32]

  const int e    = blockIdx.y;
  const int cntE = cnt[e];
  const int m0   = blockIdx.x * TM;
  if (m0 >= cntE) return;
  const int z0   = blockIdx.z * FC;

  const int tid  = threadIdx.x;
  const int wave = tid >> 5;
  const int lane = tid & 31;
  const int half = lane >> 4;
  const int lrow = lane & 15;

  // stage routing rows
  if (tid < TM) {
    bool v = (m0 + tid) < cntE;
    tkS[tid] = v ? tokid[e * MAXT + m0 + tid] : 0;
    wrS[tid] = v ? tokw [e * MAXT + m0 + tid] : 0.0f;
  }
  __syncthreads();

  // stage gathered X tile [32][1024] fp32 -> bf16 (coalesced along H)
  for (int idx = tid; idx < TM * HH; idx += 256) {
    int m = idx >> 10, h = idx & (HH - 1);
    float v = (m0 + m < cntE) ? x[(size_t)tkS[m] * HH + h] : 0.0f;
    Xs[m * XP + h] = f2bf(v);
  }
  __syncthreads();

  // GEMM1 wave tiling: 2 (M) x 4 (N) wave grid over the 32x64 h tile
  const int a_m0 = (wave & 1) * 16;
  const int a_n0 = (wave >> 1) * 16;

  v8f yacc[2][8];
#pragma unroll
  for (int ms = 0; ms < 2; ++ms)
#pragma unroll
    for (int nt = 0; nt < 8; ++nt)
#pragma unroll
      for (int r = 0; r < 8; ++r) yacc[ms][nt][r] = 0.0f;

  for (int fb = 0; fb < FC / 64; ++fb) {
    const int f0 = z0 + fb * 64;

    // ---- stage W1 panel [f(64)][k(1024)] bf16, transposed, coalesced reads ----
    for (int idx = tid; idx < 64 * HH; idx += 256) {
      int f = idx & 63, k = idx >> 6;
      Ws[f * W1P + k] = f2bf(w1[(size_t)e * HH * FF + (size_t)k * FF + (f0 + f)]);
    }
    __syncthreads();

    // ---- GEMM1: h[32x64] = X[32x1024] @ W1[1024x64], barrier-free k loop ----
    v8f hacc;
#pragma unroll
    for (int r = 0; r < 8; ++r) hacc[r] = 0.0f;
    const unsigned short* apX = Xs + (a_m0 + lrow) * XP;
    const unsigned short* bpW = Ws + (a_n0 + lrow) * W1P;
    for (int k0 = 0; k0 < HH; k0 += 32) {
      v16bf a = loadA(apX + k0, half);
      v16bf b = loadB(bpW + k0, half);
      hacc = __builtin_amdgcn_wmma_f32_16x16x32_bf16(false, a, false, b,
                                                     (short)0, hacc, false, false);
    }

    // ---- bias + GELU, store h tile to LDS (C layout: row = r + 8*half) ----
    float b1v = b1[e * FF + f0 + a_n0 + lrow];
#pragma unroll
    for (int r = 0; r < 8; ++r) {
      float hv = gelu_tanh(hacc[r] + b1v);
      Hs[(a_m0 + half * 8 + r) * HP + (a_n0 + lrow)] = f2bf(hv);
    }
    __syncthreads();   // Hs ready; all waves done reading W1 panel

    // ---- stage W2 panel [h(1024)][k(64)] bf16, transposed, coalesced reads ----
    for (int idx = tid; idx < 64 * HH; idx += 256) {
      int h = idx & (HH - 1), k = idx >> 10;
      Ws[h * W2P + k] = f2bf(w2[(size_t)e * FF * HH + (size_t)(f0 + k) * HH + h]);
    }
    __syncthreads();

    // ---- GEMM2: y[32x1024] += gelu(h)[32x64] @ W2[64x1024]; wave owns 128 cols ----
#pragma unroll
    for (int kk = 0; kk < 2; ++kk) {
      const int k0 = kk * 32;
      v16bf a0 = loadA(Hs + lrow * HP + k0, half);          // rows 0..15
      v16bf a1 = loadA(Hs + (16 + lrow) * HP + k0, half);   // rows 16..31
#pragma unroll
      for (int nt = 0; nt < 8; ++nt) {
        const int n = wave * 128 + nt * 16 + lrow;
        v16bf b = loadB(Ws + n * W2P + k0, half);
        yacc[0][nt] = __builtin_amdgcn_wmma_f32_16x16x32_bf16(false, a0, false, b,
                                                              (short)0, yacc[0][nt], false, false);
        yacc[1][nt] = __builtin_amdgcn_wmma_f32_16x16x32_bf16(false, a1, false, b,
                                                              (short)0, yacc[1][nt], false, false);
      }
    }
    __syncthreads();   // done with Hs/W2 panel before next f-block
  }

  // ---- epilogue: out[tok][col] += prob * (y + b2) ----
#pragma unroll
  for (int nt = 0; nt < 8; ++nt) {
    const int col = wave * 128 + nt * 16 + lrow;
    const float b2v = b2[e * HH + col];
#pragma unroll
    for (int ms = 0; ms < 2; ++ms) {
#pragma unroll
      for (int r = 0; r < 8; ++r) {
        const int m = ms * 16 + half * 8 + r;
        if (m0 + m < cntE) {
          float val = (yacc[ms][nt][r] + b2v) * wrS[m];
          atomicAdd(out + (size_t)tkS[m] * HH + col, val);
        }
      }
    }
  }
}

// ---------------- host launcher ----------------
extern "C" void kernel_launch(void* const* d_in, const int* in_sizes, int n_in,
                              void* d_out, int out_size, void* d_ws, size_t ws_size,
                              hipStream_t stream) {
  (void)in_sizes; (void)n_in; (void)ws_size;
  const float* x       = (const float*)d_in[0];
  const float* probs   = (const float*)d_in[1];
  const int*   experts = (const int*)  d_in[2];
  const float* w1      = (const float*)d_in[3];
  const float* b1      = (const float*)d_in[4];
  const float* w2      = (const float*)d_in[5];
  const float* b2      = (const float*)d_in[6];
  float*       out     = (float*)d_out;

  // workspace: [0,256) expert counters; then token ids; then token weights
  int*   cnt   = (int*)d_ws;
  int*   tokid = (int*)((char*)d_ws + 256);
  float* tokw  = (float*)((char*)d_ws + 256 + (size_t)EE * MAXT * sizeof(int));

  hipMemsetAsync(d_out, 0, (size_t)out_size * sizeof(float), stream);
  hipMemsetAsync(d_ws, 0, 256, stream);

  moe_route<<<(KK * TT + 255) / 256, 256, 0, stream>>>(probs, experts, cnt, tokid, tokw);

  hipFuncSetAttribute((const void*)moe_ffn,
                      hipFuncAttributeMaxDynamicSharedMemorySize, SMEM_BYTES);
  dim3 grid(MAXT / TM, EE, NZ);
  moe_ffn<<<grid, 256, SMEM_BYTES, stream>>>(x, w1, b1, w2, b2,
                                             cnt, tokid, tokw, out);
}
